// AttentionHead_13975823582193
// MI455X (gfx1250) — matrix operand order repaired and checked by reading, not compiled, not run on hardware
//
#include <hip/hip_runtime.h>

typedef __attribute__((ext_vector_type(16))) __bf16 v16bf;
typedef __attribute__((ext_vector_type(8)))  float  v8f;

#define BATCH   8
#define S_LEN   2048
#define DIN     1024
#define DH      128
#define NEG_INF (-__builtin_inff())

union FragBF {
  v16bf v;
  unsigned int u[8];
};

__device__ __forceinline__ v8f zero_v8f() {
  v8f z;
#pragma unroll
  for (int i = 0; i < 8; ++i) z[i] = 0.0f;
  return z;
}
__device__ __forceinline__ v8f wmma_bf16(const FragBF& a, const FragBF& b, v8f c) {
  return __builtin_amdgcn_wmma_f32_16x16x32_bf16(false, a.v, false, b.v,
                                                 (short)0, c, false, false);
}

// ---------------------------------------------------------------------------
// Kernel 0: one-time weight convert + transpose.
//   Wt[p][n][k] = (bf16) W_p[k][n]     p in {q,k,v}, k<DIN, n<DH
// Transposed layout makes every WMMA B-fragment dword a contiguous bf16 pair.
// ---------------------------------------------------------------------------
__global__ __launch_bounds__(256) void cvt_w_kernel(
    const float* __restrict__ Wq, const float* __restrict__ Wk,
    const float* __restrict__ Wv, __bf16* __restrict__ Wt) {
  const int i = blockIdx.x * 256 + threadIdx.x;      // 3*DIN*DH/4 threads
  const int perW = DIN * DH / 4;
  const int p   = i / perW;
  const int rem = i % perW;
  const int k   = rem / (DH / 4);
  const int n4  = (rem % (DH / 4)) * 4;
  const float* src = (p == 0) ? Wq : (p == 1) ? Wk : Wv;
  float4 f = *(const float4*)(src + (size_t)k * DH + n4);
  __bf16* d = Wt + ((size_t)p * DH + n4) * DIN + k;
  d[0 * DIN] = (__bf16)f.x;
  d[1 * DIN] = (__bf16)f.y;
  d[2 * DIN] = (__bf16)f.z;
  d[3 * DIN] = (__bf16)f.w;
}

// ---------------------------------------------------------------------------
// Kernel 1: fused QKV projection. Block = 8 waves handles one 16-row tile of
// the flattened [B*S, DIN] input:
//   - 256 threads stage the 16x1024 input tile into LDS as bf16 (once).
//   - wave w computes output n-tile w for ALL THREE projections, sharing the
//     LDS A-fragments (3 WMMAs per 32-wide k-step, A amortized 3x, input
//     loaded/converted exactly once chip-wide).
// Q,K stored bf16 row-major [B*S][DH]; V stored transposed bf16 [B][DH][S].
// ---------------------------------------------------------------------------
__global__ __launch_bounds__(256) void qkv_proj_kernel(
    const float* __restrict__ x, const __bf16* __restrict__ Wt,
    const float* __restrict__ bq, const float* __restrict__ bk,
    const float* __restrict__ bv, __bf16* __restrict__ Qb,
    __bf16* __restrict__ Kb, __bf16* __restrict__ Vt) {
  __shared__ __bf16 xs[16][DIN];  // 32 KB staged input tile

  const int tid  = threadIdx.x;
  const int row0 = blockIdx.x * 16;

  // ---- cooperative f32 -> bf16 staging of the A tile ----
  {
    const int r  = tid & 15;
    const int c0 = (tid >> 4) * 64;
    const float* src = x + (size_t)(row0 + r) * DIN + c0;
    __bf16* dst = &xs[r][c0];
#pragma unroll
    for (int j = 0; j < 64; j += 4) {
      float4 f = *(const float4*)(src + j);
      dst[j + 0] = (__bf16)f.x;
      dst[j + 1] = (__bf16)f.y;
      dst[j + 2] = (__bf16)f.z;
      dst[j + 3] = (__bf16)f.w;
    }
  }
  __syncthreads();

  const int lane = tid & 31;
  const int wv   = tid >> 5;
  const int lrow = lane & 15;
  const int half = lane >> 4;
  const int ng   = wv * 16 + lrow;  // output column for B/C fragments

  const __bf16* wq = Wt + ((size_t)0 * DH + ng) * DIN;
  const __bf16* wk = Wt + ((size_t)1 * DH + ng) * DIN;
  const __bf16* wvp = Wt + ((size_t)2 * DH + ng) * DIN;

  v8f cq = zero_v8f(), ck = zero_v8f(), cv = zero_v8f();
#pragma unroll 4
  for (int ks = 0; ks < DIN / 32; ++ks) {
    const int kb = ks * 32;
    FragBF a, fq, fk, fv;
#pragma unroll
    for (int j = 0; j < 8; ++j) {
      int col = kb + (half ? 16 : 0) + 2 * j;
      a.u[j] = *(const unsigned int*)&xs[lrow][col];
    }
    const int wb = kb + (half ? 16 : 0);
#pragma unroll
    for (int j = 0; j < 8; ++j) {
      fq.u[j] = *(const unsigned int*)(wq + wb + 2 * j);
      fk.u[j] = *(const unsigned int*)(wk + wb + 2 * j);
      fv.u[j] = *(const unsigned int*)(wvp + wb + 2 * j);
    }
    cq = wmma_bf16(a, fq, cq);
    ck = wmma_bf16(a, fk, ck);
    cv = wmma_bf16(a, fv, cv);
  }

  const float biasq = bq[ng], biask = bk[ng], biasv = bv[ng];
#pragma unroll
  for (int r = 0; r < 8; ++r) {
    const int grow = row0 + r + 8 * half;  // C layout: M = r (+8 hi half)
    Qb[(size_t)grow * DH + ng] = (__bf16)(cq[r] + biasq);
    Kb[(size_t)grow * DH + ng] = (__bf16)(ck[r] + biask);
    const int b = grow / S_LEN, s = grow % S_LEN;
    Vt[((size_t)b * DH + ng) * S_LEN + s] = (__bf16)(cv[r] + biasv);
  }
}

// ---------------------------------------------------------------------------
// Kernel 2: causal attention, reference-exact masking (tril zeros, then every
// exact 0 -> -inf), flash-style online softmax, dropout applied to the
// probabilities (NOT the denominator), PV via WMMA from transposed V.
// One wave per 16-row Q tile; 8 independent waves per block.
// ---------------------------------------------------------------------------
__global__ __launch_bounds__(256) void attn_kernel(
    const __bf16* __restrict__ Qb,   // bf16 [B*S][DH]
    const __bf16* __restrict__ Kb,   // bf16 [B*S][DH]
    const __bf16* __restrict__ Vt,   // bf16 [B][DH][S]
    const float* __restrict__ dmask, // f32 [B][S][S]
    float* __restrict__ out) {       // f32 [B*S][DH]
  __shared__ __bf16 Pbuf[8][16][32];  // per-wave P tile (16 rows x 32 keys)

  const int lane = threadIdx.x & 31;
  const int wv   = threadIdx.x >> 5;
  const int lrow = lane & 15;
  const int half = lane >> 4;

  const int qt  = blockIdx.x * 8 + wv;  // 0..1023 query tiles
  const int b   = qt >> 7;
  const int q0  = (qt & 127) * 16;      // row within batch
  const int gs0 = qt * 16;              // global flattened row

  // Q A-fragments (4 k-steps of 32 over DH=128), register-resident.
  FragBF qa[4];
  {
    const __bf16* qrow = Qb + (size_t)(gs0 + lrow) * DH;
#pragma unroll
    for (int ks = 0; ks < 4; ++ks)
#pragma unroll
      for (int j = 0; j < 8; ++j) {
        int col = ks * 32 + (half ? 16 : 0) + 2 * j;
        qa[ks].u[j] = *(const unsigned int*)(qrow + col);
      }
  }

  float mrow[8], lsum[8];
  v8f O[8];
#pragma unroll
  for (int r = 0; r < 8; ++r) { mrow[r] = NEG_INF; lsum[r] = 0.0f; }
#pragma unroll
  for (int nt = 0; nt < 8; ++nt) O[nt] = zero_v8f();

  const int kpmax = (q0 + 15) >> 5;
  for (int kp = 0; kp <= kpmax; ++kp) {
    const int k0 = kp * 32;

    // ---- S0 = Q @ K[k0..k0+15]^T, S1 = Q @ K[k0+16..k0+31]^T ----
    v8f S0 = zero_v8f(), S1 = zero_v8f();
#pragma unroll
    for (int ks = 0; ks < 4; ++ks) {
      FragBF kb0, kb1;
      const __bf16* kr0 = Kb + (size_t)(b * S_LEN + k0 + lrow) * DH;
      const __bf16* kr1 = kr0 + 16 * DH;
#pragma unroll
      for (int j = 0; j < 8; ++j) {
        int f = ks * 32 + (half ? 16 : 0) + 2 * j;
        kb0.u[j] = *(const unsigned int*)(kr0 + f);
        kb1.u[j] = *(const unsigned int*)(kr1 + f);
      }
      S0 = wmma_bf16(qa[ks], kb0, S0);
      S1 = wmma_bf16(qa[ks], kb1, S1);
    }

    // ---- reference masking semantics ----
#pragma unroll
    for (int r = 0; r < 8; ++r) {
      int qi  = q0 + r + 8 * half;
      int kj0 = k0 + lrow;
      float s0 = S0[r];
      float s1 = S1[r];
      if (kj0 > qi      || s0 == 0.0f) s0 = NEG_INF;
      if (kj0 + 16 > qi || s1 == 0.0f) s1 = NEG_INF;
      S0[r] = s0; S1[r] = s1;
    }

    // ---- online softmax + dropout; P tile to LDS as bf16 ----
#pragma unroll
    for (int r = 0; r < 8; ++r) {
      float tmax = fmaxf(S0[r], S1[r]);
      tmax = fmaxf(tmax, __shfl_xor(tmax, 1, 32));
      tmax = fmaxf(tmax, __shfl_xor(tmax, 2, 32));
      tmax = fmaxf(tmax, __shfl_xor(tmax, 4, 32));
      tmax = fmaxf(tmax, __shfl_xor(tmax, 8, 32));
      float mold = mrow[r];
      float mnew = fmaxf(mold, tmax);
      float scale = (mold == mnew) ? 1.0f : __expf(mold - mnew);
      float e0 = (S0[r] == NEG_INF) ? 0.0f : __expf(S0[r] - mnew);
      float e1 = (S1[r] == NEG_INF) ? 0.0f : __expf(S1[r] - mnew);
      float ps = e0 + e1;
      ps += __shfl_xor(ps, 1, 32);
      ps += __shfl_xor(ps, 2, 32);
      ps += __shfl_xor(ps, 4, 32);
      ps += __shfl_xor(ps, 8, 32);
      lsum[r] = lsum[r] * scale + ps;  // denominator excludes dropout
      mrow[r] = mnew;
#pragma unroll
      for (int nt = 0; nt < 8; ++nt) O[nt][r] *= scale;

      int qi  = q0 + r + 8 * half;
      int kj0 = k0 + lrow;
      size_t mb = ((size_t)b * S_LEN + qi) * S_LEN;
      float pm0 = 0.0f, pm1 = 0.0f;
      if (kj0 <= qi)      pm0 = e0 * dmask[mb + kj0];
      if (kj0 + 16 <= qi) pm1 = e1 * dmask[mb + kj0 + 16];
      Pbuf[wv][r + 8 * half][lrow]      = (__bf16)pm0;
      Pbuf[wv][r + 8 * half][lrow + 16] = (__bf16)pm1;
    }

    // ---- re-stripe P (C layout) -> A-fragment layout via LDS ----
    FragBF pa;
#pragma unroll
    for (int j = 0; j < 8; ++j) {
      int col = (half ? 16 : 0) + 2 * j;
      pa.u[j] = *(const unsigned int*)&Pbuf[wv][lrow][col];
    }

    // ---- O += P(16x32) @ V(32xDH); B-frags contiguous from transposed V ----
#pragma unroll
    for (int nt = 0; nt < 8; ++nt) {
      FragBF vb;
      const __bf16* vr =
          Vt + ((size_t)b * DH + nt * 16 + lrow) * S_LEN + k0 + (half ? 16 : 0);
#pragma unroll
      for (int j = 0; j < 8; ++j)
        vb.u[j] = *(const unsigned int*)(vr + 2 * j);
      O[nt] = wmma_bf16(pa, vb, O[nt]);
    }
  }

  // ---- normalize and store fp32 ----
#pragma unroll
  for (int r = 0; r < 8; ++r) {
    float inv = 1.0f / lsum[r];
    int grow = gs0 + r + 8 * half;
#pragma unroll
    for (int nt = 0; nt < 8; ++nt)
      out[(size_t)grow * DH + nt * 16 + lrow] = O[nt][r] * inv;
  }
}

// ---------------------------------------------------------------------------
extern "C" void kernel_launch(void* const* d_in, const int* in_sizes, int n_in,
                              void* d_out, int out_size, void* d_ws, size_t ws_size,
                              hipStream_t stream) {
  const float* x  = (const float*)d_in[0];
  const float* Wq = (const float*)d_in[1];
  const float* bq = (const float*)d_in[2];
  const float* Wk = (const float*)d_in[3];
  const float* bk = (const float*)d_in[4];
  const float* Wv = (const float*)d_in[5];
  const float* bv = (const float*)d_in[6];
  const float* dm = (const float*)d_in[7];
  float* out = (float*)d_out;

  // workspace: Qb (4MB) | Kb (4MB) | Vt (4MB) | Wt (768KB), all bf16
  __bf16* Qb = (__bf16*)d_ws;
  __bf16* Kb = Qb + (size_t)BATCH * S_LEN * DH;
  __bf16* Vt = Kb + (size_t)BATCH * S_LEN * DH;
  __bf16* Wt = Vt + (size_t)BATCH * S_LEN * DH;

  dim3 blk(256);
  cvt_w_kernel<<<dim3(3 * DIN * DH / 4 / 256), blk, 0, stream>>>(Wq, Wk, Wv, Wt);
  qkv_proj_kernel<<<dim3(BATCH * S_LEN / 16), blk, 0, stream>>>(
      x, Wt, bq, bk, bv, Qb, Kb, Vt);
  attn_kernel<<<dim3(BATCH * S_LEN / 16 / 8), blk, 0, stream>>>(
      Qb, Kb, Vt, dm, out);
}
